// VARGR_AGREE3_20091857010787
// MI455X (gfx1250) — matrix operand order repaired and checked by reading, not compiled, not run on hardware
//
#include <hip/hip_runtime.h>

typedef _Float16 f16;
typedef __attribute__((ext_vector_type(16))) f16   v16h;
typedef __attribute__((ext_vector_type(8)))  f16   v8h;
typedef __attribute__((ext_vector_type(8)))  float v8f;

// Workspace layout (f16 element offsets): transposed (N-major) weight tables.
#define OFF_E1   0        // [96][64]   relu(gz@W_e1)
#define OFF_E2   6144     // [128][96]  enc = h1@W_e2
#define OFF_A1U  18432    // [16][64]   user half of W_a1
#define OFF_A1I  19456    // [16][64]   item half of W_a1
#define OFF_P1   20480    // [16][192]  W_p1 (cols 8..15 zero-padded)
#define WS_ELEMS 23552

__device__ __forceinline__ v8f wmma16(v16h a, v16h b, v8f c) {
  return __builtin_amdgcn_wmma_f32_16x16x32_f16(false, a, false, b, (short)0, c,
                                                false, false);
}

__device__ __forceinline__ v16h zero_v16h() {
  v16h z;
#pragma unroll
  for (int i = 0; i < 16; ++i) z[i] = (f16)0.0f;
  return z;
}

// A-fragment (16x32 f16, M x K): lane<16 row=lane holds K kb+0..7 and kb+16..23,
// lane>=16 same row, K kb+8..15 and kb+24..31.
__device__ __forceinline__ v16h a_frag_lds(const f16* base, int ldk, int kb, int lane) {
  int m = lane & 15, hi = (lane >> 4) & 1;
  const f16* p = base + m * ldk + kb + hi * 8;
  v8h lo = *(const v8h*)p;
  v8h hh = *(const v8h*)(p + 16);
  return __builtin_shufflevector(lo, hh, 0,1,2,3,4,5,6,7,8,9,10,11,12,13,14,15);
}

// Same A layout but sourced from an f32 row (L2-resident embedding row),
// converting to f16 in registers. p points at row + kb + hi*8.
__device__ __forceinline__ v16h a_frag_f32(const float* p) {
  v16h v;
#pragma unroll
  for (int i = 0; i < 8; ++i) v[i] = (f16)p[i];
#pragma unroll
  for (int i = 0; i < 8; ++i) v[8 + i] = (f16)p[16 + i];
  return v;
}

// B-fragment (32x16 f16, K x N) from N-major transposed table:
// lane<16: column N=lane, K kb+0..7 and kb+16..23 ; lane>=16: column N=lane-16, K kb+8..15 and kb+24..31.
__device__ __forceinline__ v16h b_frag(const f16* bt, int ldk, int kb, int lane) {
  int n = lane & 15, hi = (lane >> 4) & 1;
  const f16* p = bt + n * ldk + kb + hi * 8;
  v8h lo = *(const v8h*)p;
  v8h hh = *(const v8h*)(p + 16);
  return __builtin_shufflevector(lo, hh, 0,1,2,3,4,5,6,7,8,9,10,11,12,13,14,15);
}

// ---------------- weight transpose / f16 pack ----------------
__global__ void vargr_prep_weights(const float* __restrict__ We1,
                                   const float* __restrict__ We2,
                                   const float* __restrict__ Wa1,
                                   const float* __restrict__ Wp1,
                                   f16* __restrict__ ws) {
  int t = blockIdx.x * 256 + threadIdx.x;
  if (t < OFF_E2) {                       // W_e1 (64,96) -> [96][64]
    int n = t / 64, k = t % 64;
    ws[t] = (f16)We1[k * 96 + n];
  } else if (t < OFF_A1U) {               // W_e2 (96,128) -> [128][96]
    int u = t - OFF_E2; int n = u / 96, k = u % 96;
    ws[t] = (f16)We2[k * 128 + n];
  } else if (t < OFF_A1I) {               // W_a1[:64] (128,16) -> [16][64]
    int u = t - OFF_A1U; int n = u / 64, k = u % 64;
    ws[t] = (f16)Wa1[k * 16 + n];
  } else if (t < OFF_P1) {                // W_a1[64:] -> [16][64]
    int u = t - OFF_A1I; int n = u / 64, k = u % 64;
    ws[t] = (f16)Wa1[(64 + k) * 16 + n];
  } else if (t < WS_ELEMS) {              // W_p1 (192,8) -> [16][192], pad n>=8
    int u = t - OFF_P1; int n = u / 192, k = u % 192;
    ws[t] = (n < 8) ? (f16)Wp1[k * 8 + n] : (f16)0.0f;
  }
}

// ---------------- fused model kernel: 16 samples per block ----------------
__global__ __launch_bounds__(128)
void vargr_fused(const float* __restrict__ user_emb,
                 const float* __restrict__ item_emb,
                 const float* __restrict__ be1g, const float* __restrict__ be2g,
                 const float* __restrict__ ba1g, const float* __restrict__ wa2g,
                 const float* __restrict__ ba2g, const float* __restrict__ bp1g,
                 const float* __restrict__ wp2g, const float* __restrict__ bp2g,
                 const float* __restrict__ eps,
                 const int* __restrict__ members,
                 const unsigned char* __restrict__ mask,   // jnp bool_ = 1 byte
                 const int* __restrict__ item_inputs,
                 const f16* __restrict__ wsT,
                 float* __restrict__ out, int B) {
  const int tid  = threadIdx.x;
  const int lane = tid & 31;
  const int wid  = tid >> 5;
  const int s0   = blockIdx.x * 16;

  __shared__ int   sidx[16 * 50];
  __shared__ int   cnti[16];
  __shared__ float cntf[16];
  __shared__ int   iidx[16];
  __shared__ f16   gzh [16 * 64];
  __shared__ f16   h1h [16 * 96];
  __shared__ f16   itemh[16 * 64];
  __shared__ float itemf[16 * 64];
  __shared__ float zmu [16 * 64];
  __shared__ float sig [16 * 64];
  __shared__ float titem[16 * 16];
  __shared__ float wsc [16 * 50];       // scores, then softmax weights
  __shared__ f16   ach [16 * 192];      // NCF input [g*it | g | it]
  __shared__ float sbe1[96], sbe2[128], sba1[16], swa2[16], sbp1[8], swp2[8];

  // ---- P1: indices, counts, biases; prefetch gathered rows into caches ----
  for (int j = tid; j < 800; j += 128) {
    int bs = s0 + j / 50; if (bs >= B) bs = B - 1;
    int u = members[bs * 50 + (j % 50)];
    sidx[j] = u;
    __builtin_prefetch(&user_emb[(size_t)u * 64], 0, 1);  // global_prefetch_b8
  }
  if (tid < 16) {
    int bs = s0 + tid; if (bs >= B) bs = B - 1;
    int c = 0;
    const unsigned char* mp = mask + (size_t)bs * 50;
    for (int m = 0; m < 50; ++m) c += mp[m];
    cnti[tid] = c; cntf[tid] = (float)c;
    iidx[tid] = item_inputs[bs];
    sba1[tid] = ba1g[tid];
    swa2[tid] = wa2g[tid];
  }
  if (tid < 96)  sbe1[tid] = be1g[tid];
  if (tid < 128) sbe2[tid] = be2g[tid];
  if (tid < 8) { sbp1[tid] = bp1g[tid]; swp2[tid] = wp2g[tid]; }
  __syncthreads();

  // ---- P2: masked mean -> relu -> gz (f16); item row (f32+f16) ----
  for (int j = tid; j < 1024; j += 128) {
    int s = j >> 6, d = j & 63;
    int c = cnti[s];
    const int* ip = &sidx[s * 50];
    float acc = 0.f;
    for (int m = 0; m < c; ++m)
      acc += user_emb[(size_t)ip[m] * 64 + d];
    gzh[s * 64 + d] = (f16)fmaxf(acc / cntf[s], 0.f);
    float iv = item_emb[(size_t)iidx[s] * 64 + d];
    itemf[s * 64 + d] = iv;
    itemh[s * 64 + d] = (f16)iv;
  }
  __syncthreads();

  // ---- P3: enc layer 1 (16x64 @ 64x96) + item attn transform (wave 3) ----
  for (int j = wid; j < 6; j += 4) {
    v8f c = {};
#pragma unroll
    for (int kb = 0; kb < 64; kb += 32)
      c = wmma16(a_frag_lds(gzh, 64, kb, lane), b_frag(wsT + OFF_E1, 64, kb, lane), c);
    int n = lane & 15, hi = lane >> 4;
#pragma unroll
    for (int rr = 0; rr < 8; ++rr) {
      int M = rr + 8 * hi, col = j * 16 + n;
      h1h[M * 96 + col] = (f16)fmaxf(c[rr] + sbe1[col], 0.f);
    }
  }
  if (wid == 3) {   // t_item = item @ W_a1[64:] + b_a1 (shared across members)
    v8f c = {};
#pragma unroll
    for (int kb = 0; kb < 64; kb += 32)
      c = wmma16(a_frag_lds(itemh, 64, kb, lane), b_frag(wsT + OFF_A1I, 64, kb, lane), c);
    int n = lane & 15, hi = lane >> 4;
#pragma unroll
    for (int rr = 0; rr < 8; ++rr) {
      int M = rr + 8 * hi;
      titem[M * 16 + n] = c[rr] + sba1[n];
    }
  }
  __syncthreads();

  // ---- P4: enc layer 2 (16x96 @ 96x128) -> z_mu, z_sigma ----
  for (int j = wid; j < 8; j += 4) {
    v8f c = {};
#pragma unroll
    for (int kb = 0; kb < 96; kb += 32)
      c = wmma16(a_frag_lds(h1h, 96, kb, lane), b_frag(wsT + OFF_E2, 96, kb, lane), c);
    int n = lane & 15, hi = lane >> 4;
#pragma unroll
    for (int rr = 0; rr < 8; ++rr) {
      int M = rr + 8 * hi, col = j * 16 + n;
      float v = c[rr] + sbe2[col];
      if (col < 64) zmu[M * 64 + col] = v;
      else          sig[M * 64 + (col - 64)] = 0.1f + 0.9f / (1.f + __expf(-v));
    }
  }

  // ---- P5: attention scores. 800 (sample,member) rows -> 50 WMMA tiles.
  //     s[b,m] = relu(u@W_a1u + t_item[b]) @ W_a2 + b_a2
  {
    float ba2v = ba2g[0];
    float wa2n = swa2[lane & 15];
    for (int t = wid; t < 50; t += 4) {
      int m16 = lane & 15, hi = lane >> 4;
      int r = t * 16 + m16;
      int s = r / 50, m = r % 50;
      const float* row = user_emb + (size_t)sidx[s * 50 + m] * 64;
      bool avalid = m < cnti[s];
      v8f c = {};
#pragma unroll
      for (int kb = 0; kb < 64; kb += 32) {
        v16h a = a_frag_f32(row + kb + hi * 8);
        if (!avalid) a = zero_v16h();
        c = wmma16(a, b_frag(wsT + OFF_A1U, 64, kb, lane), c);
      }
#pragma unroll
      for (int rr = 0; rr < 8; ++rr) {
        int M = rr + 8 * hi;
        int rg = t * 16 + M;
        int ss = rg / 50, mm = rg % 50;
        float v = fmaxf(c[rr] + titem[ss * 16 + (lane & 15)], 0.f) * wa2n;
#pragma unroll
        for (int off = 1; off < 16; off <<= 1) v += __shfl_xor(v, off, 16);
        if ((lane & 15) == 0)
          wsc[ss * 50 + mm] = (mm < cnti[ss]) ? (v + ba2v) : -3.4028235e38f;
      }
    }
  }
  __syncthreads();

  // ---- P6: softmax over members (8 lanes per sample) ----
  {
    int s = tid >> 3, la = tid & 7;
    float* sp = &wsc[s * 50];
    float mx = -3.4028235e38f;
    for (int m = la; m < 50; m += 8) mx = fmaxf(mx, sp[m]);
#pragma unroll
    for (int off = 1; off < 8; off <<= 1) mx = fmaxf(mx, __shfl_xor(mx, off, 8));
    float ex[7]; int k = 0; float sum = 0.f;
    for (int m = la; m < 50; m += 8) { float e = __expf(sp[m] - mx); ex[k++] = e; sum += e; }
#pragma unroll
    for (int off = 1; off < 8; off <<= 1) sum += __shfl_xor(sum, off, 8);
    float inv = 1.f / sum;
    k = 0;
    for (int m = la; m < 50; m += 8) sp[m] = ex[k++] * inv;
  }
  __syncthreads();

  // ---- P7: g_att (weighted gather-sum), g, build NCF input [g*it|g|it] ----
  for (int j = tid; j < 1024; j += 128) {
    int s = j >> 6, d = j & 63;
    int c = cnti[s];
    const int*   ip = &sidx[s * 50];
    const float* wp = &wsc[s * 50];
    float acc = 0.f;
    for (int m = 0; m < c; ++m)
      acc += wp[m] * user_emb[(size_t)ip[m] * 64 + d];
    int bs = s0 + s; if (bs >= B) bs = B - 1;
    float g  = acc + zmu[s * 64 + d] + sig[s * 64 + d] * eps[(size_t)bs * 64 + d];
    float it = itemf[s * 64 + d];
    ach[s * 192 + d]        = (f16)(g * it);
    ach[s * 192 + 64 + d]   = (f16)g;
    ach[s * 192 + 128 + d]  = (f16)it;
  }
  __syncthreads();

  // ---- P8: NCF head (16x192 @ 192x8 padded to 16) -> sigmoid -> out ----
  if (wid == 0) {
    v8f c = {};
#pragma unroll
    for (int kb = 0; kb < 192; kb += 32)
      c = wmma16(a_frag_lds(ach, 192, kb, lane), b_frag(wsT + OFF_P1, 192, kb, lane), c);
    int n = lane & 15, hi = lane >> 4;
    float bp2v = bp2g[0];
#pragma unroll
    for (int rr = 0; rr < 8; ++rr) {
      float v = (n < 8) ? fmaxf(c[rr] + sbp1[n], 0.f) * swp2[n] : 0.f;
#pragma unroll
      for (int off = 1; off < 16; off <<= 1) v += __shfl_xor(v, off, 16);
      if (n == 0) {
        int M = rr + 8 * hi;
        if (s0 + M < B) out[s0 + M] = 1.f / (1.f + __expf(-(v + bp2v)));
      }
    }
  }
}

extern "C" void kernel_launch(void* const* d_in, const int* in_sizes, int n_in,
                              void* d_out, int out_size, void* d_ws, size_t ws_size,
                              hipStream_t stream) {
  const float* user_emb = (const float*)d_in[0];
  const float* item_emb = (const float*)d_in[1];
  const float* W_e1 = (const float*)d_in[2];
  const float* b_e1 = (const float*)d_in[3];
  const float* W_e2 = (const float*)d_in[4];
  const float* b_e2 = (const float*)d_in[5];
  const float* W_a1 = (const float*)d_in[6];
  const float* b_a1 = (const float*)d_in[7];
  const float* W_a2 = (const float*)d_in[8];
  const float* b_a2 = (const float*)d_in[9];
  const float* W_p1 = (const float*)d_in[10];
  const float* b_p1 = (const float*)d_in[11];
  const float* W_p2 = (const float*)d_in[12];
  const float* b_p2 = (const float*)d_in[13];
  const float* eps  = (const float*)d_in[14];
  const int*   members     = (const int*)d_in[15];
  const unsigned char* msk = (const unsigned char*)d_in[16];
  const int*   item_inputs = (const int*)d_in[17];

  int B = in_sizes[17];
  f16* wsT = (f16*)d_ws;

  vargr_prep_weights<<<(WS_ELEMS + 255) / 256, 256, 0, stream>>>(
      W_e1, W_e2, W_a1, W_p1, wsT);

  int grid = (B + 15) / 16;
  vargr_fused<<<grid, 128, 0, stream>>>(
      user_emb, item_emb, b_e1, b_e2, b_a1, W_a2, b_a2, b_p1, W_p2, b_p2,
      eps, members, msk, item_inputs, wsT, (float*)d_out, B);
}
